// CutlassDynamicNeRF_11261404250467
// MI455X (gfx1250) — matrix operand order repaired and checked by reading, not compiled, not run on hardware
//
#include <hip/hip_runtime.h>
#include <hip/hip_bf16.h>
#include <math.h>

typedef __attribute__((ext_vector_type(16))) _Float16 v16h;
typedef __attribute__((ext_vector_type(8)))  _Float16 v8h;
typedef __attribute__((ext_vector_type(8)))  float    v8f;

#define BLOCK 64                 /* 2 waves */
#define ROWS_PER_BLOCK 32        /* 2 M-tiles of 16 rows, shared by both waves */

// block-shared LDS (in halves)
#define BUFIN_H (32 * 352)       /* input / concat buffer, 32 rows x 352 cols */
#define BUFH_H  (32 * 256)       /* hidden ping-pong, 32 rows x 256 cols      */
#define LDS_H   (BUFIN_H + 2 * BUFH_H)   /* 27648 halves = 55296 B */

// ---------------------------------------------------------------------------
// Prep: f32 weight matrix (Kin x Nout, row-major) -> f16 WMMA B-fragments.
// Fragment f = nt*NKT + kt is a 32x16 (KxN) tile; element layout matches the
// A fragments (contiguous k order, not AMD's table order):
//   lane: col N = lane&15, half-group h = lane>>4 ; element j -> K = 16*h + j
// Each lane's 16 halves are stored contiguously (32B) for 1-load fragments.
// ---------------------------------------------------------------------------
__global__ void nerf_prep_kernel(const float* __restrict__ src,
                                 _Float16* __restrict__ dst,
                                 int Kin, int Nout, int NKT, int NNT)
{
    int total = NKT * NNT * 512;
    for (int idx = blockIdx.x * blockDim.x + threadIdx.x; idx < total;
         idx += (int)(gridDim.x * blockDim.x)) {
        int frag = idx >> 9;
        int e    = idx & 511;
        int lane = e >> 4;
        int j    = e & 15;
        int nt   = frag / NKT;
        int kt   = frag - nt * NKT;
        int n    = nt * 16 + (lane & 15);
        int h    = lane >> 4;
        int k    = kt * 32 + 16 * h + j;
        float v  = (k < Kin && n < Nout) ? src[(size_t)k * (size_t)Nout + n] : 0.0f;
        dst[idx] = (_Float16)v;
    }
}

// ---------------------------------------------------------------------------
// A-fragment build from LDS (activations stored f16 row-major).
// A 16x32 fragment, contiguous k order: lane row = lane&15, h = lane>>4,
//   element j -> K = 16*h + j (matches B fragment order)
// => two contiguous 16B LDS loads per lane.
// ---------------------------------------------------------------------------
template <int NKT>
__device__ __forceinline__ void load_afrags(const _Float16* a_lds, int astride,
                                            v16h* a, int lane)
{
    int r = lane & 15, h = lane >> 4;
#pragma unroll
    for (int kt = 0; kt < NKT; ++kt) {
        const _Float16* p = a_lds + r * astride + kt * 32 + 16 * h;
        v8h lo = *(const v8h*)p;
        v8h hi = *(const v8h*)(p + 8);
        a[kt] = __builtin_shufflevector(lo, hi, 0, 1, 2, 3, 4, 5, 6, 7,
                                                8, 9, 10, 11, 12, 13, 14, 15);
    }
}

__device__ __forceinline__ v8f wmma_step(const v16h& a, const v16h& b, const v8f& c)
{
    return __builtin_amdgcn_wmma_f32_16x16x32_f16(
        /*neg_a=*/false, a, /*neg_b=*/false, b,
        /*c_mod=*/(short)0, c, /*reuse_a=*/false, /*reuse_b=*/false);
}

// Dense layer slice: this wave computes n-tiles [ntStart, ntEnd) for BOTH
// 16-row M-tiles, reusing each B fragment for two WMMAs.
// C/D layout: element i -> row M = i + 8*(lane>>4), col N = lane&15.
template <int NKT, bool RELU>
__device__ __forceinline__ void layer_ff(const _Float16* __restrict__ wf,
                                         const _Float16* a_lds, int astride,
                                         _Float16* o_lds, int ostride,
                                         int lane, int ntStart, int ntEnd)
{
    v16h a0[NKT], a1[NKT];
    load_afrags<NKT>(a_lds, astride, a0, lane);                  // rows  0..15
    load_afrags<NKT>(a_lds + 16 * astride, astride, a1, lane);   // rows 16..31
    const int r = lane & 15, h = lane >> 4;
    for (int nt = ntStart; nt < ntEnd; ++nt) {
        v8f acc0 = {}, acc1 = {};
#pragma unroll
        for (int kt = 0; kt < NKT; ++kt) {
            v16h b = *(const v16h*)(wf + (size_t)(nt * NKT + kt) * 512 + lane * 16);
            acc0 = wmma_step(a0[kt], b, acc0);
            acc1 = wmma_step(a1[kt], b, acc1);
        }
#pragma unroll
        for (int i = 0; i < 8; ++i) {
            float v0 = acc0[i], v1 = acc1[i];
            if (RELU) { v0 = fmaxf(v0, 0.0f); v1 = fmaxf(v1, 0.0f); }
            o_lds[(i + 8 * h) * ostride + nt * 16 + r]        = (_Float16)v0;
            o_lds[(16 + i + 8 * h) * ostride + nt * 16 + r]   = (_Float16)v1;
        }
    }
}

// ---------------------------------------------------------------------------
// Main kernel: 64 threads = 2 waves; block owns 32 rays; waves split n-tiles.
// ---------------------------------------------------------------------------
__global__ __launch_bounds__(BLOCK)
void nerf_main_kernel(const float* __restrict__ x,
                      const _Float16* __restrict__ wfrag,
                      float* __restrict__ out)
{
    __shared__ __align__(16) _Float16 smem[LDS_H];

    const int tid  = threadIdx.x;
    const int wave = tid >> 5;
    const int lane = tid & 31;
    _Float16* bufIN = smem;                 // 32 x 352
    _Float16* bufH1 = bufIN + BUFIN_H;      // 32 x 256
    _Float16* bufH2 = bufH1 + BUFH_H;       // 32 x 256
    const int rowBase = blockIdx.x * ROWS_PER_BLOCK;
    const int ntLo = wave * 8, ntHi = ntLo + 8;   // n-tile slice for 16-tile layers

    // ---- stage 0: wave0 encodes all 32 rows (lane == row) ----
    float vx = 0.f, vy = 0.f, vz = 0.f;
    if (tid < 32) {
        const int g = rowBase + tid;
        const float* xr = x + (size_t)g * 7;
        float pos[4] = {xr[0], xr[1], xr[2], xr[3]};
        vx = xr[4]; vy = xr[5]; vz = xr[6];
        _Float16* row = bufIN + tid * 352;
#pragma unroll
        for (int d = 0; d < 4; ++d) {
            float fr = 3.14159265358979f;
#pragma unroll
            for (int f = 0; f < 10; ++f) {
                float ang = pos[d] * fr;
                row[d * 20 + 2 * f]     = (_Float16)__sinf(ang);
                row[d * 20 + 2 * f + 1] = (_Float16)__cosf(ang);
                fr *= 2.0f;
            }
        }
#pragma unroll
        for (int c = 80; c < 96; ++c)   row[c] = (_Float16)0.0f;  // K-pad d1 input
#pragma unroll
        for (int c = 336; c < 352; ++c) row[c] = (_Float16)0.0f;  // K-pad d2 input
    }
    __syncthreads();

    // fragment bases (in halves; each fragment = 512 halves)
    const _Float16* W1 = wfrag + (size_t)   0 * 512;  // d1_w1  3x16
    const _Float16* W2 = wfrag + (size_t)  48 * 512;  // d1_w2  8x16
    const _Float16* W3 = wfrag + (size_t) 176 * 512;  // d1_w3  8x16
    const _Float16* W4 = wfrag + (size_t) 304 * 512;  // d2_w1 11x16
    const _Float16* W5 = wfrag + (size_t) 480 * 512;  // d2_w2  8x16
    const _Float16* W6 = wfrag + (size_t) 608 * 512;  // d2_w3  8x16
    const _Float16* W7 = wfrag + (size_t) 736 * 512;  // d2_w4  8x17
    const _Float16* W8 = wfrag + (size_t) 872 * 512;  // c_w1   9x16
    const _Float16* W9 = wfrag + (size_t)1016 * 512;  // c_w2   8x1

    // ---- density MLP part 1 ----
    layer_ff<3, true >(W1, bufIN, 352, bufH1, 256, lane, ntLo, ntHi);
    __syncthreads();
    layer_ff<8, true >(W2, bufH1, 256, bufH2, 256, lane, ntLo, ntHi);
    __syncthreads();
    // part1 (no relu) into bufIN cols 80..335 -> bufIN becomes the d2 input
    layer_ff<8, false>(W3, bufH2, 256, bufIN + 80, 352, lane, ntLo, ntHi);
    __syncthreads();

    // ---- density MLP part 2 ----
    layer_ff<11, true>(W4, bufIN, 352, bufH1, 256, lane, ntLo, ntHi);
    __syncthreads();

    // bufIN is now dead -> reuse as color-input buffer:
    //   cols 0..23 = enc_view, cols 24..279 = feature (from d2_w4), 280..287 = 0
    if (tid < 32) {
        _Float16* row = bufIN + tid * 352;
        float vv[3] = {vx, vy, vz};
#pragma unroll
        for (int d = 0; d < 3; ++d) {
            float fr = 3.14159265358979f;
#pragma unroll
            for (int f = 0; f < 4; ++f) {
                float ang = vv[d] * fr;
                row[d * 8 + 2 * f]     = (_Float16)__sinf(ang);
                row[d * 8 + 2 * f + 1] = (_Float16)__cosf(ang);
                fr *= 2.0f;
            }
        }
#pragma unroll
        for (int c = 280; c < 288; ++c) row[c] = (_Float16)0.0f;
    }

    layer_ff<8, true>(W5, bufH1, 256, bufH2, 256, lane, ntLo, ntHi);
    __syncthreads();
    layer_ff<8, true>(W6, bufH2, 256, bufH1, 256, lane, ntLo, ntHi);
    __syncthreads();

    // ---- d2_w4 (264 outs = 17 n-tiles): wave0 -> nt 0..8, wave1 -> nt 9..16 ----
    {
        v16h a0[8], a1[8];
        load_afrags<8>(bufH1, 256, a0, lane);
        load_afrags<8>(bufH1 + 16 * 256, 256, a1, lane);
        const int r = lane & 15, h = lane >> 4;

        if (wave == 0) {
            // n-tile 0: dens cols 0..15 -> heads + first 8 feature cols
            v8f acc0 = {}, acc1 = {};
#pragma unroll
            for (int kt = 0; kt < 8; ++kt) {
                v16h b = *(const v16h*)(W7 + (size_t)kt * 512 + lane * 16);
                acc0 = wmma_step(a0[kt], b, acc0);
                acc1 = wmma_step(a1[kt], b, acc1);
            }
#pragma unroll
            for (int mt = 0; mt < 2; ++mt) {
#pragma unroll
                for (int i = 0; i < 8; ++i) {
                    float v = (mt == 0) ? acc0[i] : acc1[i];
                    int M = mt * 16 + i + 8 * h;
                    size_t g = (size_t)(rowBase + M);
                    if (r < 6) {
                        out[g * 12 + 4 + r] = tanhf(v);                        // scene_flow
                    } else if (r < 8) {
                        out[g * 12 + 10 + (r - 6)] = 1.0f / (1.0f + __expf(-v)); // disocc
                    } else {
                        bufIN[M * 352 + r + 16] = (_Float16)v;                 // feature
                        if (r == 8) out[g * 12 + 3] = v;                       // density
                    }
                }
            }
        }
        const int fLo = (wave == 0) ? 1 : 9;
        const int fHi = (wave == 0) ? 9 : 17;
        for (int nt = fLo; nt < fHi; ++nt) {
            v8f acc0 = {}, acc1 = {};
#pragma unroll
            for (int kt = 0; kt < 8; ++kt) {
                v16h b = *(const v16h*)(W7 + (size_t)(nt * 8 + kt) * 512 + lane * 16);
                acc0 = wmma_step(a0[kt], b, acc0);
                acc1 = wmma_step(a1[kt], b, acc1);
            }
#pragma unroll
            for (int i = 0; i < 8; ++i) {
                int M0 = i + 8 * h;
                // dens col c = 16*nt + r  ->  color-input col c + 16
                bufIN[M0 * 352 + nt * 16 + r + 16]        = (_Float16)acc0[i];
                bufIN[(16 + M0) * 352 + nt * 16 + r + 16] = (_Float16)acc1[i];
            }
        }
    }
    __syncthreads();

    // ---- color MLP ----
    layer_ff<9, true>(W8, bufIN, 352, bufH2, 256, lane, ntLo, ntHi);
    __syncthreads();

    if (wave == 0) {   // c_w2: single n-tile, wave-uniform branch (EXEC all-1)
        v16h a0[8], a1[8];
        load_afrags<8>(bufH2, 256, a0, lane);
        load_afrags<8>(bufH2 + 16 * 256, 256, a1, lane);
        v8f acc0 = {}, acc1 = {};
#pragma unroll
        for (int kt = 0; kt < 8; ++kt) {
            v16h b = *(const v16h*)(W9 + (size_t)kt * 512 + lane * 16);
            acc0 = wmma_step(a0[kt], b, acc0);
            acc1 = wmma_step(a1[kt], b, acc1);
        }
        const int r = lane & 15, h = lane >> 4;
        if (r < 3) {
#pragma unroll
            for (int i = 0; i < 8; ++i) {
                size_t g0 = (size_t)(rowBase + i + 8 * h);
                out[g0 * 12 + r]        = acc0[i];        // rgb rows  0..15
                out[(g0 + 16) * 12 + r] = acc1[i];        // rgb rows 16..31
            }
        }
    }
}

// ---------------------------------------------------------------------------
extern "C" void kernel_launch(void* const* d_in, const int* in_sizes, int n_in,
                              void* d_out, int out_size, void* d_ws, size_t ws_size,
                              hipStream_t stream)
{
    const float* x = (const float*)d_in[0];
    _Float16* wf   = (_Float16*)d_ws;   // needs 1024 frags * 1024 B = 1 MB

    struct LayerDesc { int inIdx, Kin, Nout, NKT, NNT, fragBase; };
    const LayerDesc L[9] = {
        {1,  80, 256,  3, 16,    0},  // d1_w1
        {2, 256, 256,  8, 16,   48},  // d1_w2
        {3, 256, 256,  8, 16,  176},  // d1_w3
        {4, 336, 256, 11, 16,  304},  // d2_w1
        {5, 256, 256,  8, 16,  480},  // d2_w2
        {6, 256, 256,  8, 16,  608},  // d2_w3
        {7, 256, 264,  8, 17,  736},  // d2_w4
        {8, 280, 256,  9, 16,  872},  // c_w1
        {9, 256,   3,  8,  1, 1016},  // c_w2
    };

    for (int i = 0; i < 9; ++i) {
        int total = L[i].NKT * L[i].NNT * 512;
        int blocks = (total + 255) / 256;
        nerf_prep_kernel<<<blocks, 256, 0, stream>>>(
            (const float*)d_in[L[i].inIdx], wf + (size_t)L[i].fragBase * 512,
            L[i].Kin, L[i].Nout, L[i].NKT, L[i].NNT);
    }

    const int N = in_sizes[0] / 7;                 // 524288
    const int grid = N / ROWS_PER_BLOCK;           // 16384 blocks of 64 threads
    nerf_main_kernel<<<grid, BLOCK, 0, stream>>>(x, wf, (float*)d_out);
}
